// DiTLayer_67319317397777
// MI455X (gfx1250) — matrix operand order repaired and checked by reading, not compiled, not run on hardware
//
#include <hip/hip_runtime.h>
#include <hip/hip_bf16.h>
#include <math.h>

#define NN   32768
#define EE   524288
#define FF   128
#define HH   8
#define FMH  512
#define SIXF 768

typedef __bf16 bf16_t;
typedef bf16_t v16bf __attribute__((ext_vector_type(16)));
typedef float  v8f   __attribute__((ext_vector_type(8)));

// ---- float <-> ordered-int encoding for atomic max on floats ----
__device__ __forceinline__ int enc_f(float f){ int i=__float_as_int(f); return i>=0 ? i : (i ^ 0x7fffffff); }
__device__ __forceinline__ float dec_f(int i){ return __int_as_float(i>=0 ? i : (i ^ 0x7fffffff)); }

// ---- WMMA A-operand swizzle: element (row, k) -> fragment-contiguous LDS index ----
// Forward layout (contiguous k = 16*hi + p, 16-bit A 16x32): lane = row + hi*16, fragment elem p.
__device__ __forceinline__ int a_swz_idx(int row, int k){
  const int kk = k >> 5;
  const int km = k & 31;
  const int hi = (km >> 4) & 1;
  const int p  = km & 15;
  return ((kk << 5) + (hi << 4) + row) * 16 + p;
}

// A fragment: one contiguous 32B vector from swizzled LDS
__device__ __forceinline__ v16bf load_a_swz(const bf16_t* A, int lane, int kk){
  return *(const v16bf*)(A + (((kk << 5) + lane) << 4));
}

// B fragment: one contiguous 32B vector from pre-transposed bf16 weight WT[col][k]
__device__ __forceinline__ v16bf load_b_t(const bf16_t* __restrict__ WT, int ldk, int col, int lane, int kbase){
  return *(const v16bf*)(WT + (size_t)col * ldk + kbase + ((lane >> 4) & 1) * 16);
}

__device__ __forceinline__ v8f wmma_bf16(v16bf a, v16bf b, v8f c){
  return __builtin_amdgcn_wmma_f32_16x16x32_bf16(false, a, false, b, (short)0, c, false, false);
}

// ===================== kernel P: transpose+convert weights to bf16 =====================
__global__ __launch_bounds__(256)
void k_prep_w(const float* __restrict__ Wada, const float* __restrict__ Wq,
              const float* __restrict__ Wk,   const float* __restrict__ Wv,
              const float* __restrict__ Wkp,  const float* __restrict__ Wvp,
              const float* __restrict__ Wo,   const float* __restrict__ W1,
              const float* __restrict__ W2,
              bf16_t* __restrict__ TWada, bf16_t* __restrict__ TWq,
              bf16_t* __restrict__ TWk,   bf16_t* __restrict__ TWv,
              bf16_t* __restrict__ TWkp,  bf16_t* __restrict__ TWvp,
              bf16_t* __restrict__ TWo,   bf16_t* __restrict__ TW1,
              bf16_t* __restrict__ TW2)
{
  const int t = blockIdx.x * 256 + threadIdx.x;
  if (t < FF * SIXF){            // Wada: [128][768] -> TWada[col][k]
    const int k = t / SIXF, col = t % SIXF;
    TWada[(size_t)col * FF + k] = (bf16_t)Wada[t];
  }
  if (t < FF * FF){              // square weights [128][128] -> WT[col][k]
    const int k = t >> 7, col = t & 127;
    const size_t o = (size_t)col * FF + k;
    TWq[o]  = (bf16_t)Wq[t];  TWk[o]  = (bf16_t)Wk[t];  TWv[o] = (bf16_t)Wv[t];
    TWkp[o] = (bf16_t)Wkp[t]; TWvp[o] = (bf16_t)Wvp[t]; TWo[o] = (bf16_t)Wo[t];
  }
  if (t < FF * FMH){             // W1: [128][512] -> TW1[col][k], ldk=128
    const int k = t / FMH, col = t % FMH;
    TW1[(size_t)col * FF + k] = (bf16_t)W1[t];
  }
  if (t < FMH * FF){             // W2: [512][128] -> TW2[col][k], ldk=512
    const int k = t >> 7, col = t & 127;
    TW2[(size_t)col * FMH + k] = (bf16_t)W2[t];
  }
}

// ===================== kernel 0: init accumulators =====================
__global__ __launch_bounds__(256)
void k_init(int* __restrict__ mbuf, float* __restrict__ den, float* __restrict__ att,
            float* __restrict__ cuts, float* __restrict__ cnt)
{
  size_t t = (size_t)blockIdx.x * blockDim.x + threadIdx.x;
  if (t < (size_t)NN * HH) { mbuf[t] = 0x807fffff; den[t] = 0.f; }  // enc(-inf)
  if (t < (size_t)NN * FF)   att[t] = 0.f;
  if (t < (size_t)NN)      { cuts[t] = 0.f; cnt[t] = 0.f; }
}

// ===================== kernel 1: node pre (adaLN cond + qkv) =====================
__global__ __launch_bounds__(128)
void k_node_pre(const float* __restrict__ xg, const float* __restrict__ cg,
                const float* __restrict__ lnsc, const float* __restrict__ lnbi,
                const bf16_t* __restrict__ TWada, const float* __restrict__ bada,
                const bf16_t* __restrict__ TWq, const bf16_t* __restrict__ TWk,
                const bf16_t* __restrict__ TWv,
                float* __restrict__ mods, float* __restrict__ xpre,
                float* __restrict__ qw, float* __restrict__ kw, float* __restrict__ vw)
{
  __shared__ float  sX[16][FF];
  __shared__ __align__(32) bf16_t sA[16 * FF];   // WMMA-swizzled
  __shared__ float  sG1[16][FF];
  __shared__ float  sS1[16][FF];   // doubles as c-staging before GEMM phase

  const int tid = threadIdx.x;
  const int lane = tid & 31, w = tid >> 5;
  const int nb = blockIdx.x * 16;

  const float4* x4 = (const float4*)(xg + (size_t)nb * FF);
  const float4* c4 = (const float4*)(cg + (size_t)nb * FF);
  float4* sX4 = (float4*)&sX[0][0];
  float4* sC4 = (float4*)&sS1[0][0];
#pragma unroll
  for (int i = 0; i < 4; ++i){
    sX4[tid + i*128] = x4[tid + i*128];
    sC4[tid + i*128] = c4[tid + i*128];
  }
  __syncthreads();

  if (tid < 16){
    // LN(c; scale,bias) + silu -> swizzled bf16 A
    const int row = tid;
    float mu = 0.f;
    for (int j = 0; j < FF; ++j) mu += sS1[row][j];
    mu *= (1.f / FF);
    float va = 0.f;
    for (int j = 0; j < FF; ++j){ float d = sS1[row][j] - mu; va += d * d; }
    va *= (1.f / FF);
    const float rs = rsqrtf(va + 1e-6f);
    for (int j = 0; j < FF; ++j){
      float y = (sS1[row][j] - mu) * rs * lnsc[j] + lnbi[j];
      float s = y / (1.f + __expf(-y));
      sA[a_swz_idx(row, j)] = (bf16_t)s;
    }
  } else if (tid < 32){
    // LN(x) (no affine) in place
    const int row = tid - 16;
    float mu = 0.f;
    for (int j = 0; j < FF; ++j) mu += sX[row][j];
    mu *= (1.f / FF);
    float va = 0.f;
    for (int j = 0; j < FF; ++j){ float d = sX[row][j] - mu; va += d * d; }
    va *= (1.f / FF);
    const float rs = rsqrtf(va + 1e-6f);
    for (int j = 0; j < FF; ++j) sX[row][j] = (sX[row][j] - mu) * rs;
  }
  __syncthreads();

  const int c = lane & 15, hi = (lane >> 4) & 1;

  // mods[16 x 768] = silu(ln c) @ W_ada + b_ada ; 48 tiles -> 12 per wave
  for (int tt = 0; tt < 12; ++tt){
    const int tile = w * 12 + tt;
    const int cb = tile * 16, col = cb + c;
    v8f acc;
#pragma unroll
    for (int r = 0; r < 8; ++r) acc[r] = bada[col];
#pragma unroll
    for (int kk = 0; kk < 4; ++kk){
      v16bf a = load_a_swz(sA, lane, kk);
      v16bf b = load_b_t(TWada, FF, col, lane, kk * 32);
      acc = wmma_bf16(a, b, acc);
    }
#pragma unroll
    for (int r = 0; r < 8; ++r){
      const int row = r + hi * 8;
      const float v = acc[r];
      mods[(size_t)(nb + row) * SIXF + col] = v;
      if (tile < 8)       sG1[row][col]      = v;   // g1
      else if (tile < 16) sS1[row][col - FF] = v;   // s1
    }
  }
  __syncthreads();

  // x_pre = LN(x)*(1+g1)+s1; store + swizzled bf16 A
#pragma unroll
  for (int i = 0; i < 16; ++i){
    const int idx = tid + i * 128;
    const int row = idx >> 7, col = idx & 127;
    const float xp = sX[row][col] * (1.f + sG1[row][col]) + sS1[row][col];
    xpre[(size_t)(nb + row) * FF + col] = xp;
    sA[a_swz_idx(row, col)] = (bf16_t)xp;
  }
  __syncthreads();

  // q, k, v = x_pre @ {Wq,Wk,Wv}; 24 tiles -> 6 per wave
  for (int j = 0; j < 6; ++j){
    const int id = w * 6 + j;
    const int mat = id >> 3, tile = id & 7;
    const int cb = tile * 16, col = cb + c;
    const bf16_t* WT = (mat == 0) ? TWq : (mat == 1) ? TWk : TWv;
    float* O         = (mat == 0) ? qw  : (mat == 1) ? kw  : vw;
    v8f acc;
#pragma unroll
    for (int r = 0; r < 8; ++r) acc[r] = 0.f;
#pragma unroll
    for (int kk = 0; kk < 4; ++kk){
      v16bf a = load_a_swz(sA, lane, kk);
      v16bf b = load_b_t(WT, FF, col, lane, kk * 32);
      acc = wmma_bf16(a, b, acc);
    }
#pragma unroll
    for (int r = 0; r < 8; ++r){
      const int row = r + hi * 8;
      O[(size_t)(nb + row) * FF + col] = acc[r];
    }
  }
}

// ===================== kernel 2: edge logits + segment max =====================
__global__ __launch_bounds__(128)
void k_edge_logits(const float* __restrict__ eg, const bf16_t* __restrict__ TWkp,
                   const float* __restrict__ qw, const float* __restrict__ kw,
                   const int* __restrict__ src, const int* __restrict__ dst,
                   float* __restrict__ logits, int* __restrict__ mbuf)
{
  __shared__ __align__(32) bf16_t sE[4][16 * FF];   // swizzled, per wave
  __shared__ int sSrc[4][16];
  __shared__ int sDst[4][16];

  const int tid = threadIdx.x, lane = tid & 31, w = tid >> 5;
  const int eb0 = blockIdx.x * 64 + w * 16;

  // stage 16 edge rows as swizzled bf16
  const float4* e4 = (const float4*)(eg + (size_t)eb0 * FF);
#pragma unroll
  for (int i = 0; i < 16; ++i){
    const int idx4 = lane * 16 + i;           // 0..511
    const float4 vv = e4[idx4];
    const int b0 = idx4 * 4;
    const int o = a_swz_idx(b0 >> 7, b0 & 127);  // 4 consecutive k stay contiguous
    sE[w][o + 0] = (bf16_t)vv.x; sE[w][o + 1] = (bf16_t)vv.y;
    sE[w][o + 2] = (bf16_t)vv.z; sE[w][o + 3] = (bf16_t)vv.w;
  }
  if (lane < 16){ sSrc[w][lane] = src[eb0 + lane]; sDst[w][lane] = dst[eb0 + lane]; }
  __syncthreads();

  const int c = lane & 15, hi = (lane >> 4) & 1;
  for (int t = 0; t < HH; ++t){                 // head t == WMMA column tile (DH==16)
    v8f acc;
#pragma unroll
    for (int r = 0; r < 8; ++r) acc[r] = 0.f;
#pragma unroll
    for (int kk = 0; kk < 4; ++kk){
      v16bf a = load_a_swz(&sE[w][0], lane, kk);
      v16bf b = load_b_t(TWkp, FF, t * 16 + c, lane, kk * 32);
      acc = wmma_bf16(a, b, acc);               // kp tile
    }
    float pr[8];
#pragma unroll
    for (int r = 0; r < 8; ++r){
      const int row = r + hi * 8;
      const float* krow = kw + (size_t)sSrc[w][row] * FF;
      const float* qrow = qw + (size_t)sDst[w][row] * FF;
      const float kadd = acc[r] + krow[t * 16 + c];
      pr[r] = qrow[t * 16 + c] * kadd;
    }
#pragma unroll
    for (int m = 1; m < 16; m <<= 1){
#pragma unroll
      for (int r = 0; r < 8; ++r) pr[r] += __shfl_xor(pr[r], m, 16);
    }
    if (c == 0){
#pragma unroll
      for (int r = 0; r < 8; ++r){
        const int row = r + hi * 8;
        const int e = eb0 + row;
        const float lg = pr[r] * 0.25f;         // 1/sqrt(DH)
        logits[(size_t)e * HH + t] = lg;
        atomicMax(&mbuf[(size_t)sDst[w][row] * HH + t], enc_f(lg));
      }
    }
  }
}

// ===================== kernel 3: softmax weights + segment sums =====================
__global__ __launch_bounds__(256)
void k_edge_soft(const float* __restrict__ logits, const int* __restrict__ mbuf,
                 const int* __restrict__ dst, const float* __restrict__ cutoff,
                 float* __restrict__ wbuf, float* __restrict__ den,
                 float* __restrict__ cuts, float* __restrict__ cnt)
{
  const int e = blockIdx.x * 256 + threadIdx.x;
  if (e >= EE) return;
  const int d = dst[e];
  const float cu = cutoff[e];
#pragma unroll
  for (int h = 0; h < HH; ++h){
    const float m = dec_f(mbuf[(size_t)d * HH + h]);
    const float wv = __expf(logits[(size_t)e * HH + h] - m) * cu;
    wbuf[(size_t)e * HH + h] = wv;
    atomicAdd(&den[(size_t)d * HH + h], wv);
  }
  atomicAdd(&cuts[d], cu);
  atomicAdd(&cnt[d], 1.0f);
}

// ===================== kernel 4: vp + weighted scatter into att =====================
__global__ __launch_bounds__(128)
void k_edge_agg(const float* __restrict__ eg, const bf16_t* __restrict__ TWvp,
                const float* __restrict__ vw, const int* __restrict__ src,
                const int* __restrict__ dst, const float* __restrict__ wbuf,
                const float* __restrict__ den, float* __restrict__ att)
{
  __shared__ __align__(32) bf16_t sE[4][16 * FF];
  __shared__ int sSrc[4][16];
  __shared__ int sDst[4][16];
  __shared__ float sAW[4][16][HH];   // softmax weight / denom

  const int tid = threadIdx.x, lane = tid & 31, w = tid >> 5;
  const int eb0 = blockIdx.x * 64 + w * 16;

  const float4* e4 = (const float4*)(eg + (size_t)eb0 * FF);
#pragma unroll
  for (int i = 0; i < 16; ++i){
    const int idx4 = lane * 16 + i;
    const float4 vv = e4[idx4];
    const int b0 = idx4 * 4;
    const int o = a_swz_idx(b0 >> 7, b0 & 127);
    sE[w][o + 0] = (bf16_t)vv.x; sE[w][o + 1] = (bf16_t)vv.y;
    sE[w][o + 2] = (bf16_t)vv.z; sE[w][o + 3] = (bf16_t)vv.w;
  }
  if (lane < 16){ sSrc[w][lane] = src[eb0 + lane]; sDst[w][lane] = dst[eb0 + lane]; }
#pragma unroll
  for (int i = 0; i < 4; ++i){
    const int idx = lane * 4 + i;               // 0..127
    const int row = idx >> 3, h = idx & 7;
    const int e = eb0 + row;
    const int d = dst[e];
    const float dd = den[(size_t)d * HH + h];
    const float di = dd > 0.f ? 1.f / dd : 1.f;
    sAW[w][row][h] = wbuf[(size_t)e * HH + h] * di;
  }
  __syncthreads();

  const int c = lane & 15, hi = (lane >> 4) & 1;
  for (int t = 0; t < HH; ++t){
    v8f acc;
#pragma unroll
    for (int r = 0; r < 8; ++r) acc[r] = 0.f;
#pragma unroll
    for (int kk = 0; kk < 4; ++kk){
      v16bf a = load_a_swz(&sE[w][0], lane, kk);
      v16bf b = load_b_t(TWvp, FF, t * 16 + c, lane, kk * 32);
      acc = wmma_bf16(a, b, acc);               // vp tile
    }
#pragma unroll
    for (int r = 0; r < 8; ++r){
      const int row = r + hi * 8;
      const float a = sAW[w][row][t];
      const float* vrow = vw + (size_t)sSrc[w][row] * FF;
      const float o = a * (vrow[t * 16 + c] + acc[r]);
      atomicAdd(&att[(size_t)sDst[w][row] * FF + t * 16 + c], o);
    }
  }
}

// ===================== kernel 5: Wo + fallback + residual + MLP =====================
__global__ __launch_bounds__(128)
void k_node_post(const float* __restrict__ xg, const bf16_t* __restrict__ TWo,
                 const bf16_t* __restrict__ TW1, const float* __restrict__ b1,
                 const bf16_t* __restrict__ TW2, const float* __restrict__ b2,
                 const float* __restrict__ mods, const float* __restrict__ xpre,
                 const float* __restrict__ att, const float* __restrict__ cuts,
                 const float* __restrict__ cnt, float* __restrict__ out)
{
  __shared__ float  sX[16][FF];
  __shared__ float  sO[16][FF];
  __shared__ __align__(32) bf16_t sA[16 * FF];     // swizzled, K=128
  __shared__ __align__(32) bf16_t sH[16 * FMH];    // swizzled, K=512
  __shared__ float  sMu[16];
  __shared__ float  sRs[16];
  __shared__ float  sMask[16];

  const int tid = threadIdx.x, lane = tid & 31, w = tid >> 5;
  const int nb = blockIdx.x * 16;

  // stage residual x (f32) and att (swizzled bf16 A)
  const float4* x4 = (const float4*)(xg + (size_t)nb * FF);
  const float4* a4 = (const float4*)(att + (size_t)nb * FF);
  float4* sX4 = (float4*)&sX[0][0];
#pragma unroll
  for (int i = 0; i < 4; ++i){
    const int idx4 = tid + i * 128;
    sX4[idx4] = x4[idx4];
    const float4 av = a4[idx4];
    const int b0 = idx4 * 4;
    const int o = a_swz_idx(b0 >> 7, b0 & 127);
    sA[o + 0] = (bf16_t)av.x; sA[o + 1] = (bf16_t)av.y;
    sA[o + 2] = (bf16_t)av.z; sA[o + 3] = (bf16_t)av.w;
  }
  if (tid < 16){
    const int node = nb + tid;
    const float cm = cuts[node] / fmaxf(cnt[node], 1.f);
    sMask[tid] = (cm < 1e-5f) ? 1.f : 0.f;
  }
  __syncthreads();

  const int c = lane & 15, hi = (lane >> 4) & 1;

  // att @ Wo -> sO ; 8 tiles -> 2 per wave
  for (int tt = 0; tt < 2; ++tt){
    const int tile = w * 2 + tt;
    const int cb = tile * 16, col = cb + c;
    v8f acc;
#pragma unroll
    for (int r = 0; r < 8; ++r) acc[r] = 0.f;
#pragma unroll
    for (int kk = 0; kk < 4; ++kk){
      v16bf a = load_a_swz(sA, lane, kk);
      v16bf b = load_b_t(TWo, FF, col, lane, kk * 32);
      acc = wmma_bf16(a, b, acc);
    }
#pragma unroll
    for (int r = 0; r < 8; ++r){ const int row = r + hi * 8; sO[row][col] = acc[r]; }
  }
  __syncthreads();

  // x = x + where(mask, x_pre, att@Wo) * al1
#pragma unroll
  for (int i = 0; i < 16; ++i){
    const int idx = tid + i * 128, row = idx >> 7, col = idx & 127;
    const int node = nb + row;
    const float attf = (sMask[row] > 0.5f) ? xpre[(size_t)node * FF + col] : sO[row][col];
    const float al1 = mods[(size_t)node * SIXF + 2 * FF + col];
    sX[row][col] = sX[row][col] + attf * al1;
  }
  __syncthreads();

  if (tid < 16){
    const int row = tid;
    float mu = 0.f;
    for (int j = 0; j < FF; ++j) mu += sX[row][j];
    mu *= (1.f / FF);
    float va = 0.f;
    for (int j = 0; j < FF; ++j){ float d = sX[row][j] - mu; va += d * d; }
    va *= (1.f / FF);
    sMu[row] = mu;
    sRs[row] = rsqrtf(va + 1e-6f);
  }
  __syncthreads();

  // x_mlp = LN(x)*(1+g2)+s2 -> swizzled bf16 A
#pragma unroll
  for (int i = 0; i < 16; ++i){
    const int idx = tid + i * 128, row = idx >> 7, col = idx & 127;
    const int node = nb + row;
    const float g2 = mods[(size_t)node * SIXF + 3 * FF + col];
    const float s2 = mods[(size_t)node * SIXF + 4 * FF + col];
    const float xm = (sX[row][col] - sMu[row]) * sRs[row] * (1.f + g2) + s2;
    sA[a_swz_idx(row, col)] = (bf16_t)xm;
  }
  __syncthreads();

  // hidden = gelu(x_mlp @ W1 + b1), 16x512 -> 32 tiles -> 8 per wave (swizzled into sH)
  for (int tt = 0; tt < 8; ++tt){
    const int tile = w * 8 + tt;
    const int cb = tile * 16, col = cb + c;
    v8f acc;
#pragma unroll
    for (int r = 0; r < 8; ++r) acc[r] = b1[col];
#pragma unroll
    for (int kk = 0; kk < 4; ++kk){
      v16bf a = load_a_swz(sA, lane, kk);
      v16bf b = load_b_t(TW1, FF, col, lane, kk * 32);
      acc = wmma_bf16(a, b, acc);
    }
#pragma unroll
    for (int r = 0; r < 8; ++r){
      const int row = r + hi * 8;
      const float v = acc[r];
      const float g = 0.5f * v * (1.f + tanhf(0.7978845608f * (v + 0.044715f * v * v * v)));
      sH[a_swz_idx(row, col)] = (bf16_t)g;   // k index = col (0..511)
    }
  }
  __syncthreads();

  // mlp = hidden @ W2 + b2 ; out = x + mlp*al2 ; K=512 -> 16 WMMA steps
  for (int tt = 0; tt < 2; ++tt){
    const int tile = w * 2 + tt;
    const int cb = tile * 16, col = cb + c;
    v8f acc;
#pragma unroll
    for (int r = 0; r < 8; ++r) acc[r] = b2[col];
    for (int kk = 0; kk < 16; ++kk){
      v16bf a = load_a_swz(sH, lane, kk);
      v16bf b = load_b_t(TW2, FMH, col, lane, kk * 32);
      acc = wmma_bf16(a, b, acc);
    }
#pragma unroll
    for (int r = 0; r < 8; ++r){
      const int row = r + hi * 8;
      const int node = nb + row;
      const float al2 = mods[(size_t)node * SIXF + 5 * FF + col];
      out[(size_t)node * FF + col] = sX[row][col] + acc[r] * al2;
    }
  }
}

// ===================== launch =====================
extern "C" void kernel_launch(void* const* d_in, const int* in_sizes, int n_in,
                              void* d_out, int out_size, void* d_ws, size_t ws_size,
                              hipStream_t stream)
{
  (void)in_sizes; (void)n_in; (void)out_size; (void)ws_size;

  const float* xg     = (const float*)d_in[0];
  const float* eg     = (const float*)d_in[1];
  const float* cg     = (const float*)d_in[2];
  const float* cutoff = (const float*)d_in[3];
  const float* lnsc   = (const float*)d_in[4];
  const float* lnbi   = (const float*)d_in[5];
  const float* Wada   = (const float*)d_in[6];
  const float* bada   = (const float*)d_in[7];
  const float* Wq     = (const float*)d_in[8];
  const float* Wk     = (const float*)d_in[9];
  const float* Wv     = (const float*)d_in[10];
  const float* Wkp    = (const float*)d_in[11];
  const float* Wvp    = (const float*)d_in[12];
  const float* Wo     = (const float*)d_in[13];
  const float* W1     = (const float*)d_in[14];
  const float* b1     = (const float*)d_in[15];
  const float* W2     = (const float*)d_in[16];
  const float* b2     = (const float*)d_in[17];
  const int*   src    = (const int*)d_in[18];
  const int*   dst    = (const int*)d_in[19];

  float* ws = (float*)d_ws;
  size_t off = 0;
  float* mods   = ws + off; off += (size_t)NN * SIXF;
  float* xpre   = ws + off; off += (size_t)NN * FF;
  float* qw     = ws + off; off += (size_t)NN * FF;
  float* kw     = ws + off; off += (size_t)NN * FF;
  float* vw     = ws + off; off += (size_t)NN * FF;
  float* logits = ws + off; off += (size_t)EE * HH;
  float* wbuf   = ws + off; off += (size_t)EE * HH;
  float* att    = ws + off; off += (size_t)NN * FF;
  float* den    = ws + off; off += (size_t)NN * HH;
  float* cuts   = ws + off; off += (size_t)NN;
  float* cnt    = ws + off; off += (size_t)NN;
  int*   mbuf   = (int*)(ws + off); off += (size_t)NN * HH;

  // bf16 transposed weight pool (32B-aligned regions; sizes are multiples of 16)
  bf16_t* tw = (bf16_t*)(ws + off);
  bf16_t* TWada = tw; tw += (size_t)SIXF * FF;
  bf16_t* TWq   = tw; tw += (size_t)FF * FF;
  bf16_t* TWk   = tw; tw += (size_t)FF * FF;
  bf16_t* TWv   = tw; tw += (size_t)FF * FF;
  bf16_t* TWkp  = tw; tw += (size_t)FF * FF;
  bf16_t* TWvp  = tw; tw += (size_t)FF * FF;
  bf16_t* TWo   = tw; tw += (size_t)FF * FF;
  bf16_t* TW1   = tw; tw += (size_t)FMH * FF;
  bf16_t* TW2   = tw; tw += (size_t)FF * FMH;

  k_prep_w<<<(FF * SIXF + 255) / 256, 256, 0, stream>>>(
      Wada, Wq, Wk, Wv, Wkp, Wvp, Wo, W1, W2,
      TWada, TWq, TWk, TWv, TWkp, TWvp, TWo, TW1, TW2);
  k_init<<<((size_t)NN * FF + 255) / 256, 256, 0, stream>>>(mbuf, den, att, cuts, cnt);
  k_node_pre<<<NN / 16, 128, 0, stream>>>(xg, cg, lnsc, lnbi, TWada, bada, TWq, TWk, TWv,
                                          mods, xpre, qw, kw, vw);
  k_edge_logits<<<EE / 64, 128, 0, stream>>>(eg, TWkp, qw, kw, src, dst, logits, mbuf);
  k_edge_soft<<<EE / 256, 256, 0, stream>>>(logits, mbuf, dst, cutoff, wbuf, den, cuts, cnt);
  k_edge_agg<<<EE / 64, 128, 0, stream>>>(eg, TWvp, vw, src, dst, wbuf, den, att);
  k_node_post<<<NN / 16, 128, 0, stream>>>(xg, TWo, TW1, b1, TW2, b2, mods, xpre, att,
                                           cuts, cnt, (float*)d_out);
}
